// KoopmanCVAE_19816979104331
// MI455X (gfx1250) — matrix operand order repaired and checked by reading, not compiled, not run on hardware
//
#include <hip/hip_runtime.h>
#include <hip/hip_bf16.h>
#include <math.h>

// ---------------- types / WMMA helpers ----------------
typedef __attribute__((ext_vector_type(16))) _Float16 v16h;
typedef __attribute__((ext_vector_type(8)))  float    v8f;

__device__ inline v8f zero8() {
  v8f z;
#pragma unroll
  for (int i = 0; i < 8; ++i) z[i] = 0.0f;
  return z;
}

// Opaque zero offset: re-materialized by asm volatile every execution, so
// weight loads based on (ptr + opqz()) cannot be hoisted out of the time loop
// (which previously caused a 512-VGPR spill storm), while the base pointer
// keeps its global address-space provenance (=> global_load, not flat_load).
__device__ inline int opqz() {
  int x = 0;
  asm volatile("" : "+s"(x));
  return x;
}

// A tile 16x32 (MxK) f16, source row-major with given element stride.
// CDNA5 16-bit A layout: kb = 2*v + (v>=4 ? 8 : 0) + 8*(lane>>4)
__device__ inline v16h load_a16x32(const _Float16* base, int stride, int lane) {
  const int r  = lane & 15;
  const int hi = (lane >> 4) << 3; // 0 or 8
  const _Float16* p = base + r * stride;
  v16h o;
#pragma unroll
  for (int v = 0; v < 8; ++v) {
    const int kb = 2 * v + ((v >= 4) ? 8 : 0) + hi;
    o[2 * v]     = p[kb];
    o[2 * v + 1] = p[kb + 1];
  }
  return o;
}

// B tile 32x16 (KxN) from weight matrix stored row-major [N,K]: B(k,n)=W[n,k].
__device__ inline v16h load_b32x16(const _Float16* w, int stride, int lane) {
  const int n  = lane & 15;
  const int hi = (lane >> 4) << 3; // 0 or 8
  const _Float16* p = w + n * stride + hi;
  v16h o;
#pragma unroll
  for (int v = 0; v < 8; ++v) {
    const int kb = 2 * v + ((v >= 4) ? 8 : 0);
    o[2 * v]     = p[kb];
    o[2 * v + 1] = p[kb + 1];
  }
  return o;
}

__device__ inline v8f wmma_f16(v16h a, v16h b, v8f c) {
  return __builtin_amdgcn_wmma_f32_16x16x32_f16(false, a, false, b, (short)0, c,
                                                false, false);
}

// ---------------- math helpers ----------------
__device__ inline float symlogf(float x) { return copysignf(log1pf(fabsf(x)), x); }
__device__ inline float sigmoidf_(float x) { return 1.0f / (1.0f + expf(-x)); }
__device__ inline float siluf_(float x) { return x * sigmoidf_(x); }

// ---------------- constants ----------------
#define BB 1024
#define TT 256
#define SD_ 24
#define AD_ 6
#define EE 128
#define HH 256
#define MM 64
#define DT_ 0.1f
#define MU_ (-0.2f)

// ---------------- weight f32 -> f16 ----------------
__global__ void k_f32_to_f16(const float* __restrict__ src,
                             _Float16* __restrict__ dst, int n) {
  int i = blockIdx.x * 256 + threadIdx.x;
  if (i < n) dst[i] = (_Float16)src[i];
}

// ---------------- fused projection + 2-layer GRU + skill head ----------------
__global__ __launch_bounds__(256) void k_gru(
    const float* __restrict__ states, const float* __restrict__ actions,
    const float* __restrict__ proj_W, const float* __restrict__ proj_b,
    const float* __restrict__ proj_g, const float* __restrict__ proj_bt,
    const _Float16* __restrict__ Wih0h, const _Float16* __restrict__ Whh0h,
    const float* __restrict__ bih0, const float* __restrict__ bhh0,
    const _Float16* __restrict__ Wih1h, const _Float16* __restrict__ Whh1h,
    const float* __restrict__ bih1, const float* __restrict__ bhh1,
    const float* __restrict__ ln_g, const float* __restrict__ ln_b,
    const float* __restrict__ skill_W, const float* __restrict__ skill_b,
    float* __restrict__ P_hat) {
  __shared__ float    cat_s[16][32];
  __shared__ float    ptmp[16][EE];
  __shared__ _Float16 xt[16][EE];
  __shared__ float    h0f[16][HH];
  __shared__ _Float16 h0h[16][HH];
  __shared__ float    h1f[16][HH];
  __shared__ _Float16 h1h[16][HH];
  __shared__ float    rmu[16], rrs[16];

  const int tid  = threadIdx.x;
  const int lane = tid & 31;
  const int wv   = tid >> 5;
  const int b0   = blockIdx.x * 16;

  for (int i = tid; i < 16 * HH; i += 256) {
    int r = i >> 8, c = i & 255;
    h0f[r][c] = 0.0f; h0h[r][c] = (_Float16)0.0f;
    h1f[r][c] = 0.0f; h1h[r][c] = (_Float16)0.0f;
  }
  __syncthreads();

#pragma unroll 1
  for (int t = 0; t < TT; ++t) {
    // ---- symlog concat ----
    for (int i = tid; i < 16 * 30; i += 256) {
      int r = i / 30, k = i % 30;
      float v = (k < SD_)
                    ? states[((size_t)(b0 + r) * TT + t) * SD_ + k]
                    : actions[((size_t)(b0 + r) * TT + t) * AD_ + (k - SD_)];
      cat_s[r][k] = symlogf(v);
    }
    __syncthreads();
    // ---- projection ----
    for (int i = tid; i < 16 * EE; i += 256) {
      int r = i >> 7, e = i & 127;
      float acc = proj_b[e];
      const float* w = proj_W + e * 30;
#pragma unroll
      for (int k = 0; k < 30; ++k) acc += cat_s[r][k] * w[k];
      ptmp[r][e] = acc;
    }
    __syncthreads();
    if (tid < 16) {
      float mu = 0.0f;
      for (int e = 0; e < EE; ++e) mu += ptmp[tid][e];
      mu *= (1.0f / EE);
      float var = 0.0f;
      for (int e = 0; e < EE; ++e) { float d = ptmp[tid][e] - mu; var += d * d; }
      var *= (1.0f / EE);
      rmu[tid] = mu; rrs[tid] = rsqrtf(var + 1e-5f);
    }
    __syncthreads();
    for (int i = tid; i < 16 * EE; i += 256) {
      int r = i >> 7, e = i & 127;
      float v = (ptmp[r][e] - rmu[r]) * rrs[r] * proj_g[e] + proj_bt[e];
      xt[r][e] = (_Float16)siluf_(v);
    }
    __syncthreads();

    v8f gi[3][2], gh[3][2];

    // ================= layer 0 =================
#pragma unroll
    for (int g = 0; g < 3; ++g)
#pragma unroll
      for (int jt = 0; jt < 2; ++jt) { gi[g][jt] = zero8(); gh[g][jt] = zero8(); }

    {
      const _Float16* Wp = Wih0h + opqz();
#pragma unroll 1
      for (int k0 = 0; k0 < EE; k0 += 32) {
        v16h av = load_a16x32(&xt[0][k0], EE, lane);
#pragma unroll
        for (int g = 0; g < 3; ++g)
#pragma unroll
          for (int jt = 0; jt < 2; ++jt) {
            const int n0 = g * HH + wv * 32 + jt * 16;
            v16h bv = load_b32x16(Wp + (size_t)n0 * EE + k0, EE, lane);
            gi[g][jt] = wmma_f16(av, bv, gi[g][jt]);
          }
      }
    }
    {
      const _Float16* Wp = Whh0h + opqz();
#pragma unroll 1
      for (int k0 = 0; k0 < HH; k0 += 32) {
        v16h av = load_a16x32(&h0h[0][k0], HH, lane);
#pragma unroll
        for (int g = 0; g < 3; ++g)
#pragma unroll
          for (int jt = 0; jt < 2; ++jt) {
            const int n0 = g * HH + wv * 32 + jt * 16;
            v16h bv = load_b32x16(Wp + (size_t)n0 * HH + k0, HH, lane);
            gh[g][jt] = wmma_f16(av, bv, gh[g][jt]);
          }
      }
    }
    __syncthreads();
#pragma unroll
    for (int jt = 0; jt < 2; ++jt) {
      const int col = wv * 32 + jt * 16 + (lane & 15);
      const int mb  = (lane >> 4) * 8;
#pragma unroll
      for (int e = 0; e < 8; ++e) {
        const int m = mb + e;
        float ir = gi[0][jt][e] + bih0[col];
        float iz = gi[1][jt][e] + bih0[HH + col];
        float in_ = gi[2][jt][e] + bih0[2 * HH + col];
        float hr = gh[0][jt][e] + bhh0[col];
        float hz = gh[1][jt][e] + bhh0[HH + col];
        float hn = gh[2][jt][e] + bhh0[2 * HH + col];
        float r = sigmoidf_(ir + hr);
        float z = sigmoidf_(iz + hz);
        float n = tanhf(in_ + r * hn);
        float hv = (1.0f - z) * n + z * h0f[m][col];
        h0f[m][col] = hv; h0h[m][col] = (_Float16)hv;
      }
    }
    __syncthreads();

    // ================= layer 1 =================
#pragma unroll
    for (int g = 0; g < 3; ++g)
#pragma unroll
      for (int jt = 0; jt < 2; ++jt) { gi[g][jt] = zero8(); gh[g][jt] = zero8(); }

    {
      const _Float16* Wp = Wih1h + opqz();
#pragma unroll 1
      for (int k0 = 0; k0 < HH; k0 += 32) {
        v16h av = load_a16x32(&h0h[0][k0], HH, lane);
#pragma unroll
        for (int g = 0; g < 3; ++g)
#pragma unroll
          for (int jt = 0; jt < 2; ++jt) {
            const int n0 = g * HH + wv * 32 + jt * 16;
            v16h bv = load_b32x16(Wp + (size_t)n0 * HH + k0, HH, lane);
            gi[g][jt] = wmma_f16(av, bv, gi[g][jt]);
          }
      }
    }
    {
      const _Float16* Wp = Whh1h + opqz();
#pragma unroll 1
      for (int k0 = 0; k0 < HH; k0 += 32) {
        v16h av = load_a16x32(&h1h[0][k0], HH, lane);
#pragma unroll
        for (int g = 0; g < 3; ++g)
#pragma unroll
          for (int jt = 0; jt < 2; ++jt) {
            const int n0 = g * HH + wv * 32 + jt * 16;
            v16h bv = load_b32x16(Wp + (size_t)n0 * HH + k0, HH, lane);
            gh[g][jt] = wmma_f16(av, bv, gh[g][jt]);
          }
      }
    }
    __syncthreads();
#pragma unroll
    for (int jt = 0; jt < 2; ++jt) {
      const int col = wv * 32 + jt * 16 + (lane & 15);
      const int mb  = (lane >> 4) * 8;
#pragma unroll
      for (int e = 0; e < 8; ++e) {
        const int m = mb + e;
        float ir = gi[0][jt][e] + bih1[col];
        float iz = gi[1][jt][e] + bih1[HH + col];
        float in_ = gi[2][jt][e] + bih1[2 * HH + col];
        float hr = gh[0][jt][e] + bhh1[col];
        float hz = gh[1][jt][e] + bhh1[HH + col];
        float hn = gh[2][jt][e] + bhh1[2 * HH + col];
        float r = sigmoidf_(ir + hr);
        float z = sigmoidf_(iz + hz);
        float n = tanhf(in_ + r * hn);
        float hv = (1.0f - z) * n + z * h1f[m][col];
        h1f[m][col] = hv; h1h[m][col] = (_Float16)hv;
      }
    }
    __syncthreads();
  }

  // ---- final LN + skill logits + softmax ----
  if (tid < 16) {
    const int r = tid;
    float mu = 0.0f;
    for (int j = 0; j < HH; ++j) mu += h1f[r][j];
    mu *= (1.0f / HH);
    float var = 0.0f;
    for (int j = 0; j < HH; ++j) { float d = h1f[r][j] - mu; var += d * d; }
    var *= (1.0f / HH);
    float rs = rsqrtf(var + 1e-5f);
    float logit[8]; float mx = -1e30f;
    for (int s = 0; s < 8; ++s) {
      float acc = skill_b[s];
      const float* sw = skill_W + s * HH;
      for (int j = 0; j < HH; ++j) {
        float v = (h1f[r][j] - mu) * rs * ln_g[j] + ln_b[j];
        acc += v * sw[j];
      }
      logit[s] = acc; mx = fmaxf(mx, acc);
    }
    float se = 0.0f;
    for (int s = 0; s < 8; ++s) { logit[s] = expf(logit[s] - mx); se += logit[s]; }
    float inv = 1.0f / se;
    for (int s = 0; s < 8; ++s) P_hat[(b0 + r) * 8 + s] = logit[s] * inv;
  }
}

// ---------------- skill parameters: Vc, beta ----------------
__global__ void k_skillparams(const float* __restrict__ V,
                              const float* __restrict__ bdiag,
                              const float* __restrict__ bU,
                              const float* __restrict__ bVT,
                              float* __restrict__ Vc, float* __restrict__ beta) {
  __shared__ float nrm[8];
  const int tid = threadIdx.x;
  if (tid < 8) {
    float s = 0.0f;
    for (int m = 0; m < MM; ++m) { float v = V[tid * MM + m]; s += v * v; }
    nrm[tid] = sqrtf(s);
  }
  __syncthreads();
  for (int i = tid; i < 8 * MM; i += 64) {
    int s = i >> 6;
    float nn = fmaxf(nrm[s], 1e-8f);
    float dv = fmaxf(nn / 1.0f, 1.0f); // V_MAX = 1.0
    Vc[i] = V[i] / dv;
  }
  for (int i = tid; i < 8 * MM * AD_; i += 64) {
    int d = i % AD_, m = (i / AD_) % MM, s = i / (AD_ * MM);
    float acc = bdiag[i];
#pragma unroll
    for (int r = 0; r < 4; ++r)
      acc += bU[(s * MM + m) * 4 + r] * bVT[(s * 4 + r) * AD_ + d];
    beta[i] = fminf(fmaxf(acc, -0.5f), 0.5f); // BETA_MAX = 0.5
  }
}

// ---------------- per-batch mixing: v_eff, beta_eff ----------------
__global__ __launch_bounds__(256) void k_mix(const float* __restrict__ P,
                                             const float* __restrict__ Vc,
                                             const float* __restrict__ beta,
                                             float* __restrict__ beta_eff,
                                             _Float16* __restrict__ vbpack) {
  const int b = blockIdx.x * 4 + (threadIdx.x >> 6);
  const int m = threadIdx.x & 63;
  float p[8];
#pragma unroll
  for (int s = 0; s < 8; ++s) p[s] = P[b * 8 + s];
  float ve = 0.0f;
#pragma unroll
  for (int s = 0; s < 8; ++s) ve += p[s] * Vc[s * MM + m];
  vbpack[(size_t)b * 448 + m] = (_Float16)ve;
#pragma unroll
  for (int d = 0; d < AD_; ++d) {
    float be = 0.0f;
#pragma unroll
    for (int s = 0; s < 8; ++s) be += p[s] * beta[(s * MM + m) * AD_ + d];
    beta_eff[((size_t)b * MM + m) * AD_ + d] = be;
    vbpack[(size_t)b * 448 + MM + m * AD_ + d] = (_Float16)be;
  }
}

// ---------------- Koopman rollout -> zpack [B*T, 128] f16 ----------------
__global__ __launch_bounds__(256) void k_rollout(
    const float* __restrict__ actions, const float* __restrict__ z0re,
    const float* __restrict__ z0im, const float* __restrict__ omega,
    const float* __restrict__ beta_eff, _Float16* __restrict__ zpack) {
  const int b = blockIdx.x * 4 + (threadIdx.x >> 6);
  const int m = threadIdx.x & 63;
  float zr = z0re[b * MM + m], zi = z0im[b * MM + m];
  const float om = omega[m] * DT_;
  const float ca = cosf(om), sa = sinf(om);
  float bt[AD_];
#pragma unroll
  for (int d = 0; d < AD_; ++d) bt[d] = beta_eff[((size_t)b * MM + m) * AD_ + d];
  const float* act = actions + (size_t)b * TT * AD_;
  _Float16* zp = zpack + (size_t)b * TT * 128;
#pragma unroll 1
  for (int t = 0; t < TT; ++t) {
    float dot = 0.0f;
#pragma unroll
    for (int d = 0; d < AD_; ++d) dot += bt[d] * symlogf(act[t * AD_ + d]);
    const float decay = expf((MU_ + dot) * DT_);
    const float zr_n = decay * (ca * zr - sa * zi);
    const float zi_n = decay * (sa * zr + ca * zi);
    zr = zr_n; zi = zi_n;
    zp[t * 128 + m]      = (_Float16)zr;
    zp[t * 128 + MM + m] = (_Float16)zi;
  }
}

// ---------------- decoder: [576 -> 256 ->(LN,silu) 256 ->(silu) 6] ----------------
__global__ __launch_bounds__(256) void k_decoder(
    const _Float16* __restrict__ zpack, const _Float16* __restrict__ vbpack,
    const _Float16* __restrict__ W1h, const float* __restrict__ b1,
    const float* __restrict__ g1, const float* __restrict__ bt1,
    const _Float16* __restrict__ W2h, const float* __restrict__ b2,
    const float* __restrict__ W3, const float* __restrict__ b3,
    float* __restrict__ out) {
  __shared__ _Float16 xa[16][576];
  __shared__ float    h1raw[16][HH];
  __shared__ _Float16 h1h[16][HH];
  __shared__ float    h2f[16][HH];
  __shared__ float    rmu[16], rrs[16];

  const int tid  = threadIdx.x;
  const int lane = tid & 31;
  const int wv   = tid >> 5;
  const int b    = blockIdx.x >> 4;
  const int t0   = (blockIdx.x & 15) << 4;

  // stage A tile: cols 0..127 = zr|zi, 128..575 = v_eff|beta_eff (per-batch)
  const _Float16* zp = zpack + ((size_t)b * TT + t0) * 128;
  for (int i = tid; i < 16 * 128; i += 256) {
    int r = i >> 7, c = i & 127;
    xa[r][c] = zp[r * 128 + c];
  }
  const _Float16* vb = vbpack + (size_t)b * 448;
  for (int i = tid; i < 16 * 448; i += 256) {
    int r = i / 448, c = i % 448;
    xa[r][128 + c] = vb[c];
  }
  __syncthreads();

  // GEMM1: 576 -> 256
  v8f acc0 = zero8(), acc1 = zero8();
#pragma unroll 1
  for (int k0 = 0; k0 < 576; k0 += 32) {
    v16h av = load_a16x32(&xa[0][k0], 576, lane);
    {
      const int n0 = wv * 32;
      v16h bv = load_b32x16(W1h + (size_t)n0 * 576 + k0, 576, lane);
      acc0 = wmma_f16(av, bv, acc0);
    }
    {
      const int n0 = wv * 32 + 16;
      v16h bv = load_b32x16(W1h + (size_t)n0 * 576 + k0, 576, lane);
      acc1 = wmma_f16(av, bv, acc1);
    }
  }
  {
    const int mb = (lane >> 4) * 8;
    const int c0 = wv * 32 + (lane & 15);
    const int c1 = wv * 32 + 16 + (lane & 15);
#pragma unroll
    for (int e = 0; e < 8; ++e) {
      h1raw[mb + e][c0] = acc0[e] + b1[c0];
      h1raw[mb + e][c1] = acc1[e] + b1[c1];
    }
  }
  __syncthreads();
  if (tid < 16) {
    float mu = 0.0f;
    for (int j = 0; j < HH; ++j) mu += h1raw[tid][j];
    mu *= (1.0f / HH);
    float var = 0.0f;
    for (int j = 0; j < HH; ++j) { float d = h1raw[tid][j] - mu; var += d * d; }
    var *= (1.0f / HH);
    rmu[tid] = mu; rrs[tid] = rsqrtf(var + 1e-5f);
  }
  __syncthreads();
  for (int i = tid; i < 16 * HH; i += 256) {
    int r = i >> 8, c = i & 255;
    float v = (h1raw[r][c] - rmu[r]) * rrs[r] * g1[c] + bt1[c];
    h1h[r][c] = (_Float16)siluf_(v);
  }
  __syncthreads();

  // GEMM2: 256 -> 256, fused silu
  acc0 = zero8(); acc1 = zero8();
#pragma unroll 1
  for (int k0 = 0; k0 < HH; k0 += 32) {
    v16h av = load_a16x32(&h1h[0][k0], HH, lane);
    {
      const int n0 = wv * 32;
      v16h bv = load_b32x16(W2h + (size_t)n0 * HH + k0, HH, lane);
      acc0 = wmma_f16(av, bv, acc0);
    }
    {
      const int n0 = wv * 32 + 16;
      v16h bv = load_b32x16(W2h + (size_t)n0 * HH + k0, HH, lane);
      acc1 = wmma_f16(av, bv, acc1);
    }
  }
  {
    const int mb = (lane >> 4) * 8;
    const int c0 = wv * 32 + (lane & 15);
    const int c1 = wv * 32 + 16 + (lane & 15);
#pragma unroll
    for (int e = 0; e < 8; ++e) {
      float v0 = acc0[e] + b2[c0];
      float v1 = acc1[e] + b2[c1];
      h2f[mb + e][c0] = siluf_(v0);
      h2f[mb + e][c1] = siluf_(v1);
    }
  }
  __syncthreads();

  // GEMM3: 256 -> 6 (VALU; output f32)
  if (tid < 96) {
    const int r = tid / 6, d = tid % 6;
    float a = b3[d];
    const float* w = W3 + d * HH;
    for (int j = 0; j < HH; ++j) a += h2f[r][j] * w[j];
    out[(((size_t)b * TT) + t0 + r) * AD_ + d] = a;
  }
}

// ---------------- host launcher ----------------
extern "C" void kernel_launch(void* const* d_in, const int* in_sizes, int n_in,
                              void* d_out, int out_size, void* d_ws,
                              size_t ws_size, hipStream_t stream) {
  (void)in_sizes; (void)n_in; (void)out_size; (void)ws_size;
  const float* states   = (const float*)d_in[0];
  const float* actions  = (const float*)d_in[1];
  const float* z0_re    = (const float*)d_in[2];
  const float* z0_im    = (const float*)d_in[3];
  const float* proj_W   = (const float*)d_in[4];
  const float* proj_b   = (const float*)d_in[5];
  const float* proj_g   = (const float*)d_in[6];
  const float* proj_bt  = (const float*)d_in[7];
  const float* gWih0    = (const float*)d_in[8];
  const float* gWhh0    = (const float*)d_in[9];
  const float* gbih0    = (const float*)d_in[10];
  const float* gbhh0    = (const float*)d_in[11];
  const float* gWih1    = (const float*)d_in[12];
  const float* gWhh1    = (const float*)d_in[13];
  const float* gbih1    = (const float*)d_in[14];
  const float* gbhh1    = (const float*)d_in[15];
  const float* ln_g     = (const float*)d_in[16];
  const float* ln_b     = (const float*)d_in[17];
  const float* skill_W  = (const float*)d_in[18];
  const float* skill_b  = (const float*)d_in[19];
  const float* V        = (const float*)d_in[20];
  const float* beta_d   = (const float*)d_in[21];
  const float* beta_U   = (const float*)d_in[22];
  const float* beta_VT  = (const float*)d_in[23];
  const float* omega    = (const float*)d_in[24];
  const float* dec_W1   = (const float*)d_in[25];
  const float* dec_b1   = (const float*)d_in[26];
  const float* dec_g1   = (const float*)d_in[27];
  const float* dec_bt1  = (const float*)d_in[28];
  const float* dec_W2   = (const float*)d_in[29];
  const float* dec_b2   = (const float*)d_in[30];
  const float* dec_W3   = (const float*)d_in[31];
  const float* dec_b3   = (const float*)d_in[32];
  float* out = (float*)d_out;

  char* ws = (char*)d_ws;
  size_t off = 0;
  auto take = [&](size_t bytes) -> char* {
    char* p = ws + off;
    off += (bytes + 255) & ~(size_t)255;
    return p;
  };

  _Float16* Wih0h = (_Float16*)take((size_t)768 * 128 * 2);
  _Float16* Whh0h = (_Float16*)take((size_t)768 * 256 * 2);
  _Float16* Wih1h = (_Float16*)take((size_t)768 * 256 * 2);
  _Float16* Whh1h = (_Float16*)take((size_t)768 * 256 * 2);
  _Float16* W1h   = (_Float16*)take((size_t)256 * 576 * 2);
  _Float16* W2h   = (_Float16*)take((size_t)256 * 256 * 2);
  float*    P_hat = (float*)take((size_t)BB * 8 * 4);
  float*    Vc    = (float*)take((size_t)8 * MM * 4);
  float*    beta  = (float*)take((size_t)8 * MM * AD_ * 4);
  float*    beff  = (float*)take((size_t)BB * MM * AD_ * 4);
  _Float16* vbp   = (_Float16*)take((size_t)BB * 448 * 2);
  _Float16* zpack = (_Float16*)take((size_t)BB * TT * 128 * 2);

  auto cvt = [&](const float* s, _Float16* d, int n) {
    k_f32_to_f16<<<(n + 255) / 256, 256, 0, stream>>>(s, d, n);
  };
  cvt(gWih0, Wih0h, 768 * 128);
  cvt(gWhh0, Whh0h, 768 * 256);
  cvt(gWih1, Wih1h, 768 * 256);
  cvt(gWhh1, Whh1h, 768 * 256);
  cvt(dec_W1, W1h, 256 * 576);
  cvt(dec_W2, W2h, 256 * 256);

  k_skillparams<<<1, 64, 0, stream>>>(V, beta_d, beta_U, beta_VT, Vc, beta);

  k_gru<<<BB / 16, 256, 0, stream>>>(
      states, actions, proj_W, proj_b, proj_g, proj_bt, Wih0h, Whh0h, gbih0,
      gbhh0, Wih1h, Whh1h, gbih1, gbhh1, ln_g, ln_b, skill_W, skill_b, P_hat);

  k_mix<<<BB / 4, 256, 0, stream>>>(P_hat, Vc, beta, beff, vbp);

  k_rollout<<<BB / 4, 256, 0, stream>>>(actions, z0_re, z0_im, omega, beff,
                                        zpack);

  k_decoder<<<(BB * TT) / 16, 256, 0, stream>>>(
      zpack, vbp, W1h, dec_b1, dec_g1, dec_bt1, W2h, dec_b2, dec_W3, dec_b3,
      out);
}
